// CP_L3_sparse_outer_13975823581771
// MI455X (gfx1250) — matrix operand order repaired and checked by reading, not compiled, not run on hardware
//
#include <hip/hip_runtime.h>

// ---------------------------------------------------------------------------
// out[b,o] = (s2[b]*s3[b]) * sum_i (U1[i]*z[b,i]) * W[o,i] + bias[o]
//   s2[b] = sum_j U2[j] z[b,j],  s3[b] = sum_j U3[j] z[b,j]
// Phase 1: per-row scalars c[b] = s2*s3         (memory-bound, trivial)
// Phase 2: bf16x3 split-precision WMMA GEMM     (compute-bound, v_wmma path)
// ---------------------------------------------------------------------------

#define DIM_D 4096
#define DIM_O 4096
#define DIM_B 8192

#define BM 128
#define BN 64
#define BK 32
#define LDSTR 40   // padded LDS row stride in bf16 elements (80 bytes)

typedef float  v8f   __attribute__((ext_vector_type(8)));
typedef __bf16 v16bf __attribute__((ext_vector_type(16)));
typedef __bf16 v8bf  __attribute__((ext_vector_type(8)));
typedef __bf16 v4bf  __attribute__((ext_vector_type(4)));

// D = A*B + C, bf16 inputs, f32 accumulator
__device__ __forceinline__ v8f wmma_bf16(v16bf a, v16bf b, v8f c) {
    return __builtin_amdgcn_wmma_f32_16x16x32_bf16(
        /*neg_a=*/false, a, /*neg_b=*/false, b,
        /*c_mod=*/(short)0, c, /*reuse_a=*/false, /*reuse_b=*/false);
}

// A-fragment per CDNA5 16-bit A layout (16x32 tile):
// lanes 0-15: row=lane, K in {0..7, 16..23};
// lanes 16-31: row=lane-16, K in {8..15, 24..31}.
// kb = 8*(lane>>4); chunks at [kb, kb+8) and [kb+16, kb+24).
__device__ __forceinline__ v16bf frag_load_a(const __bf16* buf, int row, int kb) {
    const v8bf a = *reinterpret_cast<const v8bf*>(buf + row * LDSTR + kb);
    const v8bf b = *reinterpret_cast<const v8bf*>(buf + row * LDSTR + kb + 16);
    v16bf f;
#pragma unroll
    for (int i = 0; i < 8; ++i) { f[i] = a[i]; f[i + 8] = b[i]; }
    return f;
}

// B-fragment matching the A-fragment K order: lane n holds K in
// [kb, kb+8) and [kb+16, kb+24) of column n (= W row n), kb = 8*(lane>>4).
__device__ __forceinline__ v16bf frag_load_b(const __bf16* buf, int row, int kh) {
    const v8bf a = *reinterpret_cast<const v8bf*>(buf + row * LDSTR + kh);
    const v8bf b = *reinterpret_cast<const v8bf*>(buf + row * LDSTR + kh + 16);
    v16bf f;
#pragma unroll
    for (int i = 0; i < 8; ++i) { f[i] = a[i]; f[i + 8] = b[i]; }
    return f;
}

// ---------------------------------------------------------------------------
// Phase 1: c[b] = (U2 . z[b]) * (U3 . z[b]),  one wave per row
// ---------------------------------------------------------------------------
__global__ __launch_bounds__(256) void cp3_scalars_kernel(
    const float* __restrict__ z, const float* __restrict__ U2,
    const float* __restrict__ U3, float* __restrict__ c) {
    const int wid  = threadIdx.x >> 5;
    const int lane = threadIdx.x & 31;
    const int row  = blockIdx.x * 8 + wid;
    const float* zr = z + (size_t)row * DIM_D;

    float s2 = 0.f, s3 = 0.f;
#pragma unroll 4
    for (int k = lane * 4; k < DIM_D; k += 32 * 4) {
        float4 zv = *reinterpret_cast<const float4*>(zr + k);
        float4 u2 = *reinterpret_cast<const float4*>(U2 + k);
        float4 u3 = *reinterpret_cast<const float4*>(U3 + k);
        s2 += zv.x * u2.x + zv.y * u2.y + zv.z * u2.z + zv.w * u2.w;
        s3 += zv.x * u3.x + zv.y * u3.y + zv.z * u3.z + zv.w * u3.w;
    }
#pragma unroll
    for (int off = 16; off > 0; off >>= 1) {
        s2 += __shfl_xor(s2, off, 32);
        s3 += __shfl_xor(s3, off, 32);
    }
    if (lane == 0) c[row] = s2 * s3;
}

// ---------------------------------------------------------------------------
// Phase 2: GEMM  out[m,n] = c[m] * sum_k (U1[k]*z[m,k]) * W[n,k] + bias[n]
// bf16x3 split precision: x = hi + lo, acc += Ah*Bh + Ah*Bl + Al*Bh
// ---------------------------------------------------------------------------
__global__ __launch_bounds__(256) void cp3_gemm_kernel(
    const float* __restrict__ z, const float* __restrict__ U1,
    const float* __restrict__ W, const float* __restrict__ bias,
    const float* __restrict__ c, float* __restrict__ out) {

    __shared__ __bf16 Ahi[2][BM * LDSTR];
    __shared__ __bf16 Alo[2][BM * LDSTR];
    __shared__ __bf16 Bhi[2][BN * LDSTR];
    __shared__ __bf16 Blo[2][BN * LDSTR];
    __shared__ float  cLds[BM];

    const int t    = threadIdx.x;
    const int bm   = blockIdx.x;     // 0..63
    const int bn   = blockIdx.y;     // 0..63
    const int wid  = t >> 5;
    const int lane = t & 31;
    const int waveM = wid & 3;       // 4 waves over M (4*32 = 128)
    const int waveN = wid >> 2;      // 2 waves over N (2*32 = 64)

    if (t < BM) cLds[t] = c[bm * BM + t];

    // global->reg staging assignment
    const int ar = t >> 1;           // 0..127 : A row
    const int ac = (t & 1) * 16;     // 0 / 16 : A col base (16 floats per thread)
    const int br = t >> 2;           // 0..63  : B row (W row)
    const int bc = (t & 3) * 8;      // 0,8,16,24 : B col base (8 floats per thread)

    const float* zrow = z + (size_t)(bm * BM + ar) * DIM_D + ac;
    const float* wrow = W + (size_t)(bn * BN + br) * DIM_D + bc;

    float aBuf[16];
    float bBuf[8];

    auto gload = [&](int kk) {
#pragma unroll
        for (int i = 0; i < 4; ++i) {
            float4 v = *reinterpret_cast<const float4*>(zrow + kk + 4 * i);
            float4 u = *reinterpret_cast<const float4*>(U1 + kk + ac + 4 * i);
            aBuf[4 * i + 0] = v.x * u.x; aBuf[4 * i + 1] = v.y * u.y;
            aBuf[4 * i + 2] = v.z * u.z; aBuf[4 * i + 3] = v.w * u.w;
        }
#pragma unroll
        for (int i = 0; i < 2; ++i) {
            float4 v = *reinterpret_cast<const float4*>(wrow + kk + 4 * i);
            bBuf[4 * i + 0] = v.x; bBuf[4 * i + 1] = v.y;
            bBuf[4 * i + 2] = v.z; bBuf[4 * i + 3] = v.w;
        }
    };

    auto lstore = [&](int p) {
#pragma unroll
        for (int i = 0; i < 4; ++i) {
            v4bf h, l;
#pragma unroll
            for (int j = 0; j < 4; ++j) {
                float x = aBuf[4 * i + j];
                __bf16 hb = (__bf16)x;
                h[j] = hb;
                l[j] = (__bf16)(x - (float)hb);
            }
            *reinterpret_cast<v4bf*>(&Ahi[p][ar * LDSTR + ac + 4 * i]) = h;
            *reinterpret_cast<v4bf*>(&Alo[p][ar * LDSTR + ac + 4 * i]) = l;
        }
#pragma unroll
        for (int i = 0; i < 2; ++i) {
            v4bf h, l;
#pragma unroll
            for (int j = 0; j < 4; ++j) {
                float x = bBuf[4 * i + j];
                __bf16 hb = (__bf16)x;
                h[j] = hb;
                l[j] = (__bf16)(x - (float)hb);
            }
            *reinterpret_cast<v4bf*>(&Bhi[p][br * LDSTR + bc + 4 * i]) = h;
            *reinterpret_cast<v4bf*>(&Blo[p][br * LDSTR + bc + 4 * i]) = l;
        }
    };

    v8f acc[2][2];
#pragma unroll
    for (int i = 0; i < 2; ++i)
#pragma unroll
        for (int j = 0; j < 2; ++j) acc[i][j] = (v8f)(0.f);

    const int fr = lane & 15;            // fragment row within 16-tile
    const int kb = (lane >> 4) * 8;      // A: K base per lane-half (interleaved)
    const int kh = (lane >> 4) * 16;     // unused; B fragments also use kb

    gload(0);
    lstore(0);
    __syncthreads();

    const int NK = DIM_D / BK;           // 128
    for (int k = 0; k < NK; ++k) {
        const int p = k & 1;
        if (k + 1 < NK) gload((k + 1) * BK);

        const int r0 = waveM * 32 + fr;
        const int n0 = waveN * 32 + fr;
        v16bf a0h = frag_load_a(Ahi[p], r0,      kb);
        v16bf a0l = frag_load_a(Alo[p], r0,      kb);
        v16bf a1h = frag_load_a(Ahi[p], r0 + 16, kb);
        v16bf a1l = frag_load_a(Alo[p], r0 + 16, kb);
        v16bf b0h = frag_load_b(Bhi[p], n0,      kb);
        v16bf b0l = frag_load_b(Blo[p], n0,      kb);
        v16bf b1h = frag_load_b(Bhi[p], n0 + 16, kb);
        v16bf b1l = frag_load_b(Blo[p], n0 + 16, kb);

        acc[0][0] = wmma_bf16(a0h, b0h, acc[0][0]);
        acc[0][0] = wmma_bf16(a0h, b0l, acc[0][0]);
        acc[0][0] = wmma_bf16(a0l, b0h, acc[0][0]);

        acc[0][1] = wmma_bf16(a0h, b1h, acc[0][1]);
        acc[0][1] = wmma_bf16(a0h, b1l, acc[0][1]);
        acc[0][1] = wmma_bf16(a0l, b1h, acc[0][1]);

        acc[1][0] = wmma_bf16(a1h, b0h, acc[1][0]);
        acc[1][0] = wmma_bf16(a1h, b0l, acc[1][0]);
        acc[1][0] = wmma_bf16(a1l, b0h, acc[1][0]);

        acc[1][1] = wmma_bf16(a1h, b1h, acc[1][1]);
        acc[1][1] = wmma_bf16(a1h, b1l, acc[1][1]);
        acc[1][1] = wmma_bf16(a1l, b1h, acc[1][1]);

        if (k + 1 < NK) lstore((k + 1) & 1);
        __syncthreads();
    }

    // Epilogue: out = c[m]*acc + bias[n]
    // C/D layout: VGPR i -> M = i + 8*(lane>=16), N = lane&15
    const int nl = lane & 15;
    const int mh = lane >> 4;
#pragma unroll
    for (int tn = 0; tn < 2; ++tn) {
        const int ncol = bn * BN + waveN * 32 + tn * 16 + nl;
        const float bv = bias[ncol];
#pragma unroll
        for (int tm = 0; tm < 2; ++tm) {
            const int mbase = waveM * 32 + tm * 16 + mh * 8;
            v8f a = acc[tm][tn];
#pragma unroll
            for (int i = 0; i < 8; ++i) {
                const int ml = mbase + i;
                out[(size_t)(bm * BM + ml) * DIM_O + ncol] = cLds[ml] * a[i] + bv;
            }
        }
    }
}

extern "C" void kernel_launch(void* const* d_in, const int* in_sizes, int n_in,
                              void* d_out, int out_size, void* d_ws, size_t ws_size,
                              hipStream_t stream) {
    const float* z    = (const float*)d_in[0];
    const float* U1   = (const float*)d_in[1];
    const float* U2   = (const float*)d_in[2];
    const float* U3   = (const float*)d_in[3];
    const float* W    = (const float*)d_in[4];
    const float* bias = (const float*)d_in[5];
    float* out = (float*)d_out;
    float* c   = (float*)d_ws;   // DIM_B floats = 32 KB scratch

    cp3_scalars_kernel<<<DIM_B / 8, 256, 0, stream>>>(z, U2, U3, c);

    dim3 grid(DIM_B / BM, DIM_O / BN);
    cp3_gemm_kernel<<<grid, 256, 0, stream>>>(z, U1, W, bias, c, out);
}
